// RoPEAttenion_60378650247246
// MI455X (gfx1250) — matrix operand order repaired and checked by reading, not compiled, not run on hardware
//
#include <hip/hip_runtime.h>

typedef __attribute__((ext_vector_type(16))) __bf16 v16bf;
typedef __attribute__((ext_vector_type(8)))  __bf16 v8bf;
typedef __attribute__((ext_vector_type(8)))  float  v8f;

#define WMMA_BF16(A,B,C) \
  __builtin_amdgcn_wmma_f32_16x16x32_bf16(false,(A),false,(B),(short)0,(C),false,false)

#define SHUF16(LO,HI) __builtin_shufflevector((LO),(HI),0,1,2,3,4,5,6,7,8,9,10,11,12,13,14,15)

constexpr int S  = 2048;
constexpr int D  = 1024;
constexpr int H  = 16;
constexpr int DH = 64;
constexpr float ROPE_LOG = 0.28782313662425572f; // ln(10000)/32

static __device__ __forceinline__ float rowmax16(float v) {
  v = fmaxf(v, __shfl_xor(v, 1, 32));
  v = fmaxf(v, __shfl_xor(v, 2, 32));
  v = fmaxf(v, __shfl_xor(v, 4, 32));
  v = fmaxf(v, __shfl_xor(v, 8, 32));
  return v;
}
static __device__ __forceinline__ float rowsum16(float v) {
  v += __shfl_xor(v, 1, 32);
  v += __shfl_xor(v, 2, 32);
  v += __shfl_xor(v, 4, 32);
  v += __shfl_xor(v, 8, 32);
  return v;
}

// ---------------------------------------------------------------------------
// Kernel 0: pre-convert the four weight matrices fp32 -> bf16 (one pass).
// grid = (512, 4); each thread converts 8 contiguous elements.
// ---------------------------------------------------------------------------
__global__ __launch_bounds__(256) void cvt_weights_kernel(
    const float* __restrict__ Wq, const float* __restrict__ Wk,
    const float* __restrict__ Wv, const float* __restrict__ Wo,
    unsigned short* __restrict__ wc_)
{
  const float* src = (blockIdx.y == 0) ? Wq : (blockIdx.y == 1) ? Wk
                   : (blockIdx.y == 2) ? Wv : Wo;
  __bf16* dst = (__bf16*)wc_ + (size_t)blockIdx.y * D * D;
  const size_t i0 = ((size_t)blockIdx.x * 256 + threadIdx.x) * 8;
  const v8f x = *(const v8f*)(src + i0);
  v8bf o;
#pragma unroll
  for (int e = 0; e < 8; ++e) o[e] = (__bf16)x[e];
  *(v8bf*)(dst + i0) = o;
}

// ---------------------------------------------------------------------------
// Kernel 1: fused QKV projection (X @ W^T, bf16 weights) + RoPE(q,k) + q-scale.
// One wave = one 16x128 output tile (8 WMMA accumulators, A reused 8x).
// q,k -> [b,h,s,64] bf16; v -> transposed [b,h,64,s] bf16.
// ---------------------------------------------------------------------------
__global__ __launch_bounds__(256) void qkv_rope_kernel(
    const float* __restrict__ Qin, const float* __restrict__ Kin,
    const float* __restrict__ Vin,
    const unsigned short* __restrict__ wc_,
    unsigned short* __restrict__ qo_, unsigned short* __restrict__ ko_,
    unsigned short* __restrict__ vto_)
{
  __bf16* qo  = (__bf16*)qo_;
  __bf16* ko  = (__bf16*)ko_;
  __bf16* vto = (__bf16*)vto_;

  const int lane = threadIdx.x & 31;
  const int wave = threadIdx.x >> 5;
  int tile = blockIdx.x * 8 + wave;         // 6144 wave-tiles total
  const int mat = tile >> 11;               // 0=q, 1=k, 2=v  (2048 tiles each)
  tile &= 2047;
  const int row0 = (tile >> 3) * 16;        // flattened (b*S+s) row tile
  const int col0 = (tile & 7) * 128;        // output-feature group (two heads)

  const float* X  = (mat == 0) ? Qin : (mat == 1) ? Kin : Vin;
  const __bf16* W = (const __bf16*)wc_ + (size_t)mat * D * D;

  const int am  = lane & 15;                // A fragment: row
  const int akb = (lane < 16) ? 0 : 16;     // A fragment: K base
  const int cn  = lane & 15;                // B/C: column within 16
  const int bkb = (lane < 16) ? 0 : 16;     // B fragment: K base

  v8f acc[8] = {};
  const float* Xrow = X + (size_t)(row0 + am) * D;
  for (int kk = 0; kk < D; kk += 32) {
    const v8f xlo = *(const v8f*)(Xrow + kk + akb);
    const v8f xhi = *(const v8f*)(Xrow + kk + akb + 8);
    v16bf a;
#pragma unroll
    for (int e = 0; e < 8; ++e) { a[e] = (__bf16)xlo[e]; a[e + 8] = (__bf16)xhi[e]; }
#pragma unroll
    for (int j = 0; j < 8; ++j) {
      const __bf16* Wrow = W + (size_t)(col0 + j*16 + cn) * D + kk + bkb;
      const v8bf blo = *(const v8bf*)(Wrow);
      const v8bf bhi = *(const v8bf*)(Wrow + 8);
      acc[j] = WMMA_BF16(a, SHUF16(blo, bhi), acc[j]);
    }
  }

  const int rowoff = (lane < 16) ? 0 : 8;
  const int bidx  = row0 >> 11;             // / S
  const int sbase = (row0 & (S - 1)) + rowoff;

  if (mat < 2) {                            // RoPE on q and k (wave-uniform)
#pragma unroll
    for (int j = 0; j < 8; ++j) {
      const int dh = (col0 + j*16 + cn) & 63;
      const float theta = __expf(-(float)(dh >> 1) * ROPE_LOG);
      const bool even = (cn & 1) == 0;
#pragma unroll
      for (int r = 0; r < 8; ++r) {
        float sn, cs;
        __sincosf((float)(sbase + r) * theta, &sn, &cs);
        float x = acc[j][r];
        float y = __shfl_xor(x, 1, 32);     // partner of the interleaved pair
        acc[j][r] = even ? (x*cs - y*sn) : (x*cs + y*sn);
      }
    }
  }

  const float qscale = (mat == 0) ? 0.125f : 1.0f;  // 1/sqrt(64) folded into q
#pragma unroll
  for (int j = 0; j < 8; ++j) {
    const int n  = col0 + j*16 + cn;
    const int dh = n & 63;
    const int bh = bidx * H + (n >> 6);
#pragma unroll
    for (int r = 0; r < 8; ++r) {
      const int s = sbase + r;
      const __bf16 v = (__bf16)(acc[j][r] * qscale);
      if (mat == 0)      qo [((size_t)bh * S + s) * DH + dh] = v;
      else if (mat == 1) ko [((size_t)bh * S + s) * DH + dh] = v;
      else               vto[((size_t)bh * DH + dh) * S + s] = v;
    }
  }
}

// ---------------------------------------------------------------------------
// Kernel 2: causal flash attention. Grid (q-blocks=32, bh=32), 4 waves/block,
// each wave owns a 16-query tile; streams 64-key tiles with online softmax.
// ---------------------------------------------------------------------------
__global__ __launch_bounds__(128) void attn_kernel(
    const unsigned short* __restrict__ qb_, const unsigned short* __restrict__ kb_,
    const unsigned short* __restrict__ vtb_, const int* __restrict__ amask,
    unsigned short* __restrict__ ao_)
{
  const __bf16* qb  = (const __bf16*)qb_;
  const __bf16* kb  = (const __bf16*)kb_;
  const __bf16* vtb = (const __bf16*)vtb_;
  __bf16* ao = (__bf16*)ao_;

  __shared__ __bf16 psh[4][16][64];         // per-wave P staging (C->A relayout)

  const int lane = threadIdx.x & 31;
  const int wave = threadIdx.x >> 5;
  const int bh = blockIdx.y;
  const int b  = bh >> 4;
  const int h  = bh & 15;
  const int q0 = (blockIdx.x * 4 + wave) * 16;

  const int am  = lane & 15;
  const int akb = (lane < 16) ? 0 : 16;
  const int cn  = lane & 15;
  const int bkb = (lane < 16) ? 0 : 16;
  const int rowoff = (lane < 16) ? 0 : 8;

  // Q tile as two A fragments (features 0..31, 32..63); pre-scaled by 1/sqrt(dh)
  const __bf16* qp = qb + ((size_t)bh * S + q0 + am) * DH;
  v16bf qa[2];
#pragma unroll
  for (int f = 0; f < 2; ++f) {
    const v8bf lo = *(const v8bf*)(qp + f*32 + akb);
    const v8bf hi = *(const v8bf*)(qp + f*32 + akb + 8);
    qa[f] = SHUF16(lo, hi);
  }

  float mstate[8], lstate[8];
  v8f oacc[4] = {};
#pragma unroll
  for (int r = 0; r < 8; ++r) { mstate[r] = -1e30f; lstate[r] = 0.f; }

  const int qlast = q0 + 15;
  for (int kv0 = 0; kv0 <= qlast; kv0 += 64) {   // causal: skip upper triangle
    // S_tile(16x64) = Q * K^T
    v8f sj[4] = {};
#pragma unroll
    for (int t = 0; t < 4; ++t) {
      const __bf16* kp = kb + ((size_t)bh * S + kv0 + t*16 + cn) * DH;
      const v8bf k0 = *(const v8bf*)(kp + bkb);
      const v8bf k1 = *(const v8bf*)(kp + bkb + 8);
      const v8bf k2 = *(const v8bf*)(kp + 32 + bkb);
      const v8bf k3 = *(const v8bf*)(kp + 32 + bkb + 8);
      sj[t] = WMMA_BF16(qa[0], SHUF16(k0, k1), sj[t]);
      sj[t] = WMMA_BF16(qa[1], SHUF16(k2, k3), sj[t]);
    }
    // causal + padding mask
#pragma unroll
    for (int t = 0; t < 4; ++t) {
      const int key = kv0 + t*16 + cn;
      const int pok = amask[(size_t)b * S + key];
#pragma unroll
      for (int r = 0; r < 8; ++r) {
        const int qrow = q0 + rowoff + r;
        if (!pok || key > qrow) sj[t][r] = -1e30f;
      }
    }
    // online softmax update
#pragma unroll
    for (int r = 0; r < 8; ++r) {
      float v = fmaxf(fmaxf(sj[0][r], sj[1][r]), fmaxf(sj[2][r], sj[3][r]));
      v = rowmax16(v);
      const float mnew  = fmaxf(mstate[r], v);
      const float alpha = __expf(mstate[r] - mnew);
      float ps = 0.f;
#pragma unroll
      for (int t = 0; t < 4; ++t) {
        const float p = __expf(sj[t][r] - mnew);
        psh[wave][rowoff + r][t*16 + cn] = (__bf16)p;
        ps += p;
      }
      lstate[r] = lstate[r] * alpha + rowsum16(ps);
      mstate[r] = mnew;
#pragma unroll
      for (int j = 0; j < 4; ++j) oacc[j][r] *= alpha;
    }
    // reload P as A fragments (same-wave LDS ordering; compiler inserts waits)
    v16bf pa[2];
#pragma unroll
    for (int f = 0; f < 2; ++f) {
      const v8bf lo = *(const v8bf*)(&psh[wave][am][f*32 + akb]);
      const v8bf hi = *(const v8bf*)(&psh[wave][am][f*32 + akb + 8]);
      pa[f] = SHUF16(lo, hi);
    }
    // O += P * V (V stored transposed -> contiguous B fragments)
#pragma unroll
    for (int j = 0; j < 4; ++j) {
      const __bf16* vp = vtb + ((size_t)bh * DH + j*16 + cn) * S + kv0;
      const v8bf w0 = *(const v8bf*)(vp + bkb);
      const v8bf w1 = *(const v8bf*)(vp + bkb + 8);
      const v8bf w2 = *(const v8bf*)(vp + 32 + bkb);
      const v8bf w3 = *(const v8bf*)(vp + 32 + bkb + 8);
      oacc[j] = WMMA_BF16(pa[0], SHUF16(w0, w1), oacc[j]);
      oacc[j] = WMMA_BF16(pa[1], SHUF16(w2, w3), oacc[j]);
    }
  }

  // normalize and write attn output [b][s][h*64+dh] (bf16)
#pragma unroll
  for (int j = 0; j < 4; ++j) {
    const int dh = j*16 + cn;
#pragma unroll
    for (int r = 0; r < 8; ++r) {
      const int s = q0 + rowoff + r;
      const float inv = 1.0f / fmaxf(lstate[r], 1e-20f);
      ao[((size_t)b * S + s) * D + h*64 + dh] = (__bf16)(oacc[j][r] * inv);
    }
  }
}

// ---------------------------------------------------------------------------
// Kernel 3: output projection  out = attn @ Wo^T + bo  (fp32 result).
// One wave = one 16x128 tile; A and B both bf16, no converts in the loop.
// ---------------------------------------------------------------------------
__global__ __launch_bounds__(256) void out_proj_kernel(
    const unsigned short* __restrict__ ab_, const unsigned short* __restrict__ wc_,
    const float* __restrict__ bo, float* __restrict__ out)
{
  const __bf16* ab = (const __bf16*)ab_;
  const __bf16* Wc = (const __bf16*)wc_ + (size_t)3 * D * D;  // Wo slot
  const int lane = threadIdx.x & 31;
  const int wave = threadIdx.x >> 5;
  const int tile = blockIdx.x * 8 + wave;   // 2048 wave-tiles
  const int row0 = (tile >> 3) * 16;
  const int col0 = (tile & 7) * 128;

  const int am  = lane & 15;
  const int akb = (lane < 16) ? 0 : 16;
  const int cn  = lane & 15;
  const int bkb = (lane < 16) ? 0 : 16;

  v8f acc[8] = {};
  const __bf16* Arow = ab + (size_t)(row0 + am) * D;
  for (int kk = 0; kk < D; kk += 32) {
    const v8bf alo = *(const v8bf*)(Arow + kk + akb);
    const v8bf ahi = *(const v8bf*)(Arow + kk + akb + 8);
    const v16bf a = SHUF16(alo, ahi);
#pragma unroll
    for (int j = 0; j < 8; ++j) {
      const __bf16* Wrow = Wc + (size_t)(col0 + j*16 + cn) * D + kk + bkb;
      const v8bf blo = *(const v8bf*)(Wrow);
      const v8bf bhi = *(const v8bf*)(Wrow + 8);
      acc[j] = WMMA_BF16(a, SHUF16(blo, bhi), acc[j]);
    }
  }

  const int rowoff = (lane < 16) ? 0 : 8;
#pragma unroll
  for (int j = 0; j < 8; ++j) {
    const int n = col0 + j*16 + cn;
    const float bias = bo[n];
#pragma unroll
    for (int r = 0; r < 8; ++r)
      out[(size_t)(row0 + rowoff + r) * D + n] = acc[j][r] + bias;
  }
}

// ---------------------------------------------------------------------------
// Host launcher. Workspace layout (bf16, 40 MB total, fully rewritten per call):
//   wc [4*1024*1024] | q [2*16*2048*64] | k [same] | v^T [same] | attn [4096*1024]
// ---------------------------------------------------------------------------
extern "C" void kernel_launch(void* const* d_in, const int* in_sizes, int n_in,
                              void* d_out, int out_size, void* d_ws, size_t ws_size,
                              hipStream_t stream) {
  (void)in_sizes; (void)n_in; (void)out_size; (void)ws_size;
  const float* Q  = (const float*)d_in[0];
  const float* K  = (const float*)d_in[1];
  const float* V  = (const float*)d_in[2];
  const int*   am = (const int*)d_in[3];
  const float* Wq = (const float*)d_in[4];
  const float* Wk = (const float*)d_in[5];
  const float* Wv = (const float*)d_in[6];
  const float* Wo = (const float*)d_in[7];
  const float* bo = (const float*)d_in[8];
  float* out = (float*)d_out;

  const size_t wsz = (size_t)4 * D * D;     // 4M bf16 elements (all weights)
  const size_t xsz = (size_t)2 * S * D;     // 4M bf16 elements per tensor
  unsigned short* wc  = (unsigned short*)d_ws;
  unsigned short* qb  = wc  + wsz;
  unsigned short* kb  = qb  + xsz;
  unsigned short* vtb = kb  + xsz;
  unsigned short* ab  = vtb + xsz;

  // weights fp32 -> bf16 (4 x 1M elements, 8 per thread)
  cvt_weights_kernel<<<dim3(512, 4), 256, 0, stream>>>(Wq, Wk, Wv, Wo, wc);
  // 6144 wave-tiles / 8 waves per block
  qkv_rope_kernel<<<768, 256, 0, stream>>>(Q, K, V, wc, qb, kb, vtb);
  // 32 q-blocks x 32 (b,h) pairs, 4 waves each
  attn_kernel<<<dim3(32, 32), 128, 0, stream>>>(qb, kb, vtb, am, ab);
  // 2048 wave-tiles / 8 waves per block
  out_proj_kernel<<<256, 256, 0, stream>>>(ab, wc, bo, out);
}
